// GNN_32117765440105
// MI455X (gfx1250) — compile-verified, matrix operand order checked by reading, not run on hardware
//
#include <hip/hip_runtime.h>
#include <hip/hip_bf16.h>

#define DD 256
#define LL 3
#define ROWS 64   // M rows per block (4 16-row M-tiles per wave)

typedef __attribute__((ext_vector_type(16))) _Float16 v16h;
typedef __attribute__((ext_vector_type(8)))  float    v8f;
typedef __attribute__((ext_vector_type(8)))  _Float16 v8h;

// e^2 / (4*pi*eps0*1e-10)
static constexpr float COEF = 2.3070774e-18f;

__device__ __forceinline__ float mishf(float x) {
    float sp = (x > 20.0f) ? x : log1pf(__expf(x));
    return x * tanhf(sp);
}

// Load one 16x32 f16 WMMA fragment row (A or B^T layout) from a row-major
// [*, 256] f16 matrix. Per the CDNA5 ISA A-layout: halfword j of the 16-half
// fragment holds K = 16*(j/8) + 8*half + (j%8)  => two contiguous 16B chunks.
__device__ __forceinline__ v16h frag_ld(const _Float16* __restrict__ p,
                                        int row, int kt, int half) {
    const _Float16* q = p + (size_t)row * DD + kt * 32 + half * 8;
    v8h lo = *(const v8h*)(q);
    v8h hi = *(const v8h*)(q + 16);
    v16h r;
#pragma unroll
    for (int i = 0; i < 8; ++i) { r[i] = lo[i]; r[i + 8] = hi[i]; }
    return r;
}

__device__ __forceinline__ v8f zero8() {
    v8f z = {0.f, 0.f, 0.f, 0.f, 0.f, 0.f, 0.f, 0.f};
    return z;
}

// Fused 2-layer MLP over a [M,256] f16 input: out = (mish(X@W1+b1))@W2+b2
// Block: 256 threads (8 waves), 64 output rows per block, full 256 cols.
// Each wave: 4 M-tiles x 2 N-tiles -> B fragments reused 4x (L2 traffic /4).
// EPI 0: store f16                          (node src/dst branches)
// EPI 1: h_nodes = hs*hd*COEF/val -> f16    (edge branch, fused gather)
// EPI 2: atomicAdd into agg[edge_dst[row]]  (message branch, segment-sum)
template <int EPI>
__global__ __launch_bounds__(256) void mlp_wmma(
    const _Float16* __restrict__ X, int M,
    const _Float16* __restrict__ W1t, const float* __restrict__ B1,
    const _Float16* __restrict__ W2t, const float* __restrict__ B2,
    _Float16* __restrict__ out16, float* __restrict__ agg,
    const int* __restrict__ e_src, const int* __restrict__ e_dst,
    const _Float16* __restrict__ hsrc, const _Float16* __restrict__ hdst) {
    __shared__ __align__(16) _Float16 Xs[ROWS * DD];
    __shared__ __align__(16) _Float16 Ts[ROWS * DD];

    const int t    = threadIdx.x;
    const int wave = t >> 5;
    const int lane = t & 31;
    const int lm   = lane & 15;
    const int half = lane >> 4;
    const int row0 = blockIdx.x * ROWS;

    // Stage X tile: 64 rows x 512B = 32KB; 128B per thread.
    {
        int rr = t >> 2;
        int cc = (t & 3) * 64;   // in halves; 64 halves = 128B
        int gr = row0 + rr;
        if (gr >= M) gr = M - 1;
        const v8h* src = (const v8h*)(X + (size_t)gr * DD + cc);
        v8h* dst = (v8h*)(Xs + rr * DD + cc);
#pragma unroll
        for (int i = 0; i < 8; ++i) dst[i] = src[i];
    }
    __syncthreads();

    const int n0 = wave * 32;
    v8f acc[4][2];

    // ---- GEMM1: Xs @ W1 ----
#pragma unroll
    for (int mt = 0; mt < 4; ++mt) { acc[mt][0] = zero8(); acc[mt][1] = zero8(); }
#pragma unroll
    for (int kt = 0; kt < 8; ++kt) {
        v16h b0 = frag_ld(W1t, n0 + lm, kt, half);
        v16h b1 = frag_ld(W1t, n0 + 16 + lm, kt, half);
#pragma unroll
        for (int mt = 0; mt < 4; ++mt) {
            v16h a = frag_ld(Xs + mt * 16 * DD, lm, kt, half);
            acc[mt][0] = __builtin_amdgcn_wmma_f32_16x16x32_f16(
                false, a, false, b0, (short)0, acc[mt][0], false, false);
            acc[mt][1] = __builtin_amdgcn_wmma_f32_16x16x32_f16(
                false, a, false, b1, (short)0, acc[mt][1], false, false);
        }
    }
    // bias + mish -> Ts (f16)
    {
        float bb0 = B1[n0 + lm];
        float bb1 = B1[n0 + 16 + lm];
#pragma unroll
        for (int mt = 0; mt < 4; ++mt) {
#pragma unroll
            for (int v = 0; v < 8; ++v) {
                int m = mt * 16 + v + half * 8;
                Ts[m * DD + n0 + lm]      = (_Float16)mishf(acc[mt][0][v] + bb0);
                Ts[m * DD + n0 + 16 + lm] = (_Float16)mishf(acc[mt][1][v] + bb1);
            }
        }
    }
    __syncthreads();

    // ---- GEMM2: Ts @ W2 ----
#pragma unroll
    for (int mt = 0; mt < 4; ++mt) { acc[mt][0] = zero8(); acc[mt][1] = zero8(); }
#pragma unroll
    for (int kt = 0; kt < 8; ++kt) {
        v16h b0 = frag_ld(W2t, n0 + lm, kt, half);
        v16h b1 = frag_ld(W2t, n0 + 16 + lm, kt, half);
#pragma unroll
        for (int mt = 0; mt < 4; ++mt) {
            v16h a = frag_ld(Ts + mt * 16 * DD, lm, kt, half);
            acc[mt][0] = __builtin_amdgcn_wmma_f32_16x16x32_f16(
                false, a, false, b0, (short)0, acc[mt][0], false, false);
            acc[mt][1] = __builtin_amdgcn_wmma_f32_16x16x32_f16(
                false, a, false, b1, (short)0, acc[mt][1], false, false);
        }
    }

    // ---- epilogue ----
    {
        const int gc0 = n0 + lm;
        const int gc1 = n0 + 16 + lm;
        const float bb0 = B2[gc0];
        const float bb1 = B2[gc1];
#pragma unroll
        for (int mt = 0; mt < 4; ++mt) {
#pragma unroll
            for (int v = 0; v < 8; ++v) {
                int gr = row0 + mt * 16 + v + half * 8;
                if (gr >= M) continue;
                float x0 = acc[mt][0][v] + bb0;
                float x1 = acc[mt][1][v] + bb1;
                if (EPI == 0) {
                    out16[(size_t)gr * DD + gc0] = (_Float16)x0;
                    out16[(size_t)gr * DD + gc1] = (_Float16)x1;
                } else if (EPI == 1) {
                    int s = e_src[gr];
                    int d = e_dst[gr];
                    float hs0 = (float)hsrc[(size_t)s * DD + gc0];
                    float hd0 = (float)hdst[(size_t)d * DD + gc0];
                    float hs1 = (float)hsrc[(size_t)s * DD + gc1];
                    float hd1 = (float)hdst[(size_t)d * DD + gc1];
                    out16[(size_t)gr * DD + gc0] = (_Float16)(hs0 * hd0 * COEF / x0);
                    out16[(size_t)gr * DD + gc1] = (_Float16)(hs1 * hd1 * COEF / x1);
                } else {
                    int d = e_dst[gr];
                    __hip_atomic_fetch_add(&agg[(size_t)d * DD + gc0], x0,
                                           __ATOMIC_RELAXED, __HIP_MEMORY_SCOPE_AGENT);
                    __hip_atomic_fetch_add(&agg[(size_t)d * DD + gc1], x1,
                                           __ATOMIC_RELAXED, __HIP_MEMORY_SCOPE_AGENT);
                }
            }
        }
    }
}

__global__ void k_zero(float* __restrict__ p, size_t n) {
    size_t i = (size_t)blockIdx.x * 256 + threadIdx.x;
    if (i < n) p[i] = 0.0f;
}

// w: [nmats,256,256] f32 (k,n)  ->  wt: [nmats,256,256] f16 (n,k)
__global__ void k_cvt_w(const float* __restrict__ w, _Float16* __restrict__ wt,
                        int nmats) {
    size_t i = (size_t)blockIdx.x * 256 + threadIdx.x;
    size_t tot = (size_t)nmats * DD * DD;
    if (i >= tot) return;
    int k = (int)(i % DD);
    size_t r = i / DD;
    int n = (int)(r % DD);
    size_t mat = r / DD;
    wt[i] = (_Float16)w[mat * DD * DD + (size_t)k * DD + n];
}

__global__ void k_rbf(const float* __restrict__ r, _Float16* __restrict__ ef) {
    int e = blockIdx.x;
    int d = threadIdx.x;
    float x = r[(size_t)e * 3 + 0];
    float y = r[(size_t)e * 3 + 1];
    float z = r[(size_t)e * 3 + 2];
    float dist = sqrtf(x * x + y * y + z * z);
    float c = (float)d * (1.0f / (float)(DD - 1));
    float tt = dist - c;
    ef[(size_t)e * DD + d] = (_Float16)__expf(-(float)(DD - 1) * tt * tt);
}

__global__ void k_embed(const int* __restrict__ types, const float* __restrict__ af,
                        const float* __restrict__ ew, const float* __restrict__ eb,
                        float* __restrict__ h32, _Float16* __restrict__ h16) {
    int n = blockIdx.x;
    int d = threadIdx.x;
    int tpe = types[n];
    const float* a = af + (size_t)tpe * 200;
    float s = eb[d];
#pragma unroll 4
    for (int k = 0; k < 200; ++k) s += a[k] * ew[(size_t)k * DD + d];
    h32[(size_t)n * DD + d] = s;
    h16[(size_t)n * DD + d] = (_Float16)s;
}

__global__ void k_update(float* __restrict__ h32, const float* __restrict__ agg,
                         _Float16* __restrict__ h16, size_t n) {
    size_t i = (size_t)blockIdx.x * 256 + threadIdx.x;
    if (i >= n) return;
    float x = h32[i] + agg[i];
    float sp = (x > 20.0f) ? x : log1pf(__expf(x));
    h32[i] = sp;
    h16[i] = (_Float16)sp;
}

__global__ void k_pool(const float* __restrict__ h32, const int* __restrict__ gid,
                       const float* __restrict__ ow, float* __restrict__ pool,
                       float* __restrict__ cnt, int N) {
    int n = blockIdx.x * 256 + threadIdx.x;
    if (n >= N) return;
    const float* hr = h32 + (size_t)n * DD;
    float s = 0.0f;
#pragma unroll 8
    for (int d = 0; d < DD; ++d) s += hr[d] * ow[d];
    int g = gid[n];
    __hip_atomic_fetch_add(&pool[g], s, __ATOMIC_RELAXED, __HIP_MEMORY_SCOPE_AGENT);
    __hip_atomic_fetch_add(&cnt[g], 1.0f, __ATOMIC_RELAXED, __HIP_MEMORY_SCOPE_AGENT);
}

__global__ void k_final(const float* __restrict__ pool, const float* __restrict__ cnt,
                        const float* __restrict__ ob, float* __restrict__ out, int G) {
    int g = blockIdx.x * 256 + threadIdx.x;
    if (g >= G) return;
    float c = cnt[g];
    if (c < 1.0f) c = 1.0f;
    out[g] = pool[g] / c + ob[0];
}

extern "C" void kernel_launch(void* const* d_in, const int* in_sizes, int n_in,
                              void* d_out, int out_size, void* d_ws, size_t ws_size,
                              hipStream_t stream) {
    const int*   atom_types = (const int*)d_in[0];
    const int*   edge_src   = (const int*)d_in[1];
    const int*   edge_dst   = (const int*)d_in[2];
    const int*   graph_ids  = (const int*)d_in[3];
    const float* r          = (const float*)d_in[4];
    const float* af_table   = (const float*)d_in[5];
    const float* emb_w      = (const float*)d_in[6];
    const float* emb_b      = (const float*)d_in[7];
    const float* cw1        = (const float*)d_in[8];
    const float* cb1        = (const float*)d_in[9];
    const float* cw2        = (const float*)d_in[10];
    const float* cb2        = (const float*)d_in[11];
    const float* out_w      = (const float*)d_in[12];
    const float* out_b      = (const float*)d_in[13];

    const int N = in_sizes[0];
    const int E = in_sizes[1];
    const int G = out_size;

    // workspace carve-up (256B aligned)
    char* base = (char*)d_ws;
    size_t off = 0;
    auto take = [&](size_t bytes) -> char* {
        char* p = base + off;
        off += (bytes + 255) & ~(size_t)255;
        return p;
    };
    float*    h32   = (float*)take((size_t)N * DD * 4);
    _Float16* h16   = (_Float16*)take((size_t)N * DD * 2);
    _Float16* hsrc  = (_Float16*)take((size_t)N * DD * 2);
    _Float16* hdst  = (_Float16*)take((size_t)N * DD * 2);
    float*    agg   = (float*)take((size_t)N * DD * 4);
    _Float16* ef16  = (_Float16*)take((size_t)E * DD * 2);
    _Float16* hn16  = (_Float16*)take((size_t)E * DD * 2);
    _Float16* w1t   = (_Float16*)take((size_t)LL * 4 * DD * DD * 2);
    _Float16* w2t   = (_Float16*)take((size_t)LL * 4 * DD * DD * 2);
    float*    pool  = (float*)take((size_t)2 * G * 4);
    float*    cnt   = pool + G;

    const int nmats = LL * 4;
    const size_t wtot = (size_t)nmats * DD * DD;

    // preprocessing
    k_cvt_w<<<(unsigned)((wtot + 255) / 256), 256, 0, stream>>>(cw1, w1t, nmats);
    k_cvt_w<<<(unsigned)((wtot + 255) / 256), 256, 0, stream>>>(cw2, w2t, nmats);
    k_rbf<<<E, DD, 0, stream>>>(r, ef16);
    k_embed<<<N, DD, 0, stream>>>(atom_types, af_table, emb_w, emb_b, h32, h16);
    k_zero<<<(unsigned)((2 * G + 255) / 256), 256, 0, stream>>>(pool, (size_t)2 * G);

    const unsigned mb_n = (unsigned)((N + ROWS - 1) / ROWS);
    const unsigned mb_e = (unsigned)((E + ROWS - 1) / ROWS);
    const size_t nd = (size_t)N * DD;

    for (int l = 0; l < LL; ++l) {
        const size_t mo = (size_t)l * 4 * DD * DD;
        const size_t bo = (size_t)l * 4 * DD;
        // src / dst node branches
        mlp_wmma<0><<<mb_n, 256, 0, stream>>>(h16, N, w1t + mo, cb1 + bo,
                                              w2t + mo, cb2 + bo, hsrc, nullptr,
                                              nullptr, nullptr, nullptr, nullptr);
        mlp_wmma<0><<<mb_n, 256, 0, stream>>>(h16, N, w1t + mo + (size_t)DD * DD,
                                              cb1 + bo + DD, w2t + mo + (size_t)DD * DD,
                                              cb2 + bo + DD, hdst, nullptr,
                                              nullptr, nullptr, nullptr, nullptr);
        k_zero<<<(unsigned)((nd + 255) / 256), 256, 0, stream>>>(agg, nd);
        // edge branch: e_lin MLP fused with Coulomb scaling + gathers -> h_nodes
        mlp_wmma<1><<<mb_e, 256, 0, stream>>>(ef16, E, w1t + mo + 2 * (size_t)DD * DD,
                                              cb1 + bo + 2 * DD,
                                              w2t + mo + 2 * (size_t)DD * DD,
                                              cb2 + bo + 2 * DD, hn16, nullptr,
                                              edge_src, edge_dst, hsrc, hdst);
        // message MLP fused with segment-sum via fp32 atomics
        mlp_wmma<2><<<mb_e, 256, 0, stream>>>(hn16, E, w1t + mo + 3 * (size_t)DD * DD,
                                              cb1 + bo + 3 * DD,
                                              w2t + mo + 3 * (size_t)DD * DD,
                                              cb2 + bo + 3 * DD, nullptr, agg,
                                              nullptr, edge_dst, nullptr, nullptr);
        k_update<<<(unsigned)((nd + 255) / 256), 256, 0, stream>>>(h32, agg, h16, nd);
    }

    k_pool<<<(unsigned)((N + 255) / 256), 256, 0, stream>>>(h32, graph_ids, out_w,
                                                            pool, cnt, N);
    k_final<<<(unsigned)((G + 255) / 256), 256, 0, stream>>>(pool, cnt, out_b,
                                                             (float*)d_out, G);
}
